// Inter_MAE_58213986730030
// MI455X (gfx1250) — matrix operand order repaired and checked by reading, not compiled, not run on hardware
//
#include <hip/hip_runtime.h>
#include <math.h>

// ---------------- model constants ----------------
#define B_SZ   32
#define N_PTS  8192
#define G_GRP  256
#define K_NN   32
#define D_MODEL 384
#define N_VIS  103
#define M_MASK 153
#define HEADS  6
#define HDIM   64
#define M_ENC  (B_SZ * N_VIS)   // 3296
#define M_DEC  (B_SZ * G_GRP)   // 8192

// ---------------- WMMA types / helpers ----------------
typedef __attribute__((ext_vector_type(16))) __bf16 v16bf;
typedef __attribute__((ext_vector_type(8)))  float  v8f;

union BF16x16 { v16bf v; unsigned short u[16]; uint4 q[2]; };

__device__ __forceinline__ unsigned short f2bf(float f){
  unsigned int u = __float_as_uint(f);
  unsigned int r = u + 0x7FFFu + ((u >> 16) & 1u);   // RNE
  return (unsigned short)(r >> 16);
}
__device__ __forceinline__ float bf2f(unsigned short s){
  return __uint_as_float(((unsigned int)s) << 16);
}
__device__ __forceinline__ unsigned int pk2(float a, float b){
  return ((unsigned int)f2bf(b) << 16) | (unsigned int)f2bf(a);
}
// Fragment-order index helpers (within one 16x32 bf16 tile of 512 elems):
// lane holds 16 contiguous bf16 at lane*16.
__device__ __forceinline__ int fragA_idx(int row, int k){
  int kl = k & 31;
  int lane16 = (row & 15) + ((kl & 8) ? 16 : 0);
  int e = (kl & 7) | (((kl >> 4) & 1) << 3);
  return lane16 * 16 + e;
}
__device__ __forceinline__ int fragB_idx(int col, int k){
  int kl = k & 31;
  int lane16 = (col & 15) + ((kl & 8) ? 16 : 0);
  int e = (kl & 7) | (((kl >> 4) & 1) << 3);
  return lane16 * 16 + e;
}
__device__ __forceinline__ float gelu_f(float x){
  return 0.5f * x * (1.0f + erff(x * 0.70710678118654752f));
}
__device__ __forceinline__ float wave_sum(float v){
  for (int o = 16; o; o >>= 1) v += __shfl_xor(v, o, 32);
  return v;
}
__device__ __forceinline__ float wave_max(float v){
  for (int o = 16; o; o >>= 1) v = fmaxf(v, __shfl_xor(v, o, 32));
  return v;
}
__device__ __forceinline__ v8f v8f_zero(){
  v8f z;
#pragma unroll
  for (int i = 0; i < 8; i++) z[i] = 0.0f;
  return z;
}

// ---------------- weight pre-swizzle: fp32 row-major -> bf16 B-fragment order ----------------
__global__ void pack_wB_kernel(const float* __restrict__ W, int K, int N,
                               unsigned short* __restrict__ out){
  int i = blockIdx.x * blockDim.x + threadIdx.x;
  if (i >= K * N) return;
  int k = i / N, col = i % N;
  out[(((size_t)(k >> 5) * (N >> 4)) + (col >> 4)) * 512 + fragB_idx(col, k)] = f2bf(W[i]);
}

// ---------------- bf16-WMMA GEMM (LDS-free; A and B both pre-swizzled bf16) ----------------
// C[M,N] = act( A[M,K] @ B[K,N] + bias + resid )
// Af: A-fragment order bf16 [(row/16)*(K/32) + k/32][512]
// Bf: B-fragment order bf16 [(k/32)*(N/16) + col/16][512]
// C  (fp32 row-major, optional) and/or Cf (A-fragment bf16 with Kdim=N, optional).
__global__ __launch_bounds__(256) void gemm_bf16_kernel(
    const unsigned short* __restrict__ Af, const unsigned short* __restrict__ Bf,
    float* __restrict__ C, unsigned short* __restrict__ Cf,
    const float* __restrict__ bias, const float* __restrict__ resid,
    int Mr, int Nc, int Kd, int act)
{
  const int tid = threadIdx.x, lane = tid & 31, wid = tid >> 5;
  const int rb = blockIdx.y * 128, cb = blockIdx.x * 64;
  const int rowTile = blockIdx.y * 8 + wid;
  const int nkt = Kd >> 5, ntiles = Nc >> 4;
  v8f acc[4];
#pragma unroll
  for (int t = 0; t < 4; t++) acc[t] = v8f_zero();

  for (int kc = 0; kc < Kd; kc += 32){
    BF16x16 ua;
    const uint4* ap = (const uint4*)(Af + ((size_t)rowTile * nkt + (kc >> 5)) * 512 + lane * 16);
    ua.q[0] = ap[0]; ua.q[1] = ap[1];
    const size_t kbase = (size_t)(kc >> 5) * ntiles + (cb >> 4);
#pragma unroll
    for (int t = 0; t < 4; t++){
      BF16x16 ub;
      const uint4* bp = (const uint4*)(Bf + (kbase + t) * 512 + lane * 16);
      ub.q[0] = bp[0]; ub.q[1] = bp[1];
      acc[t] = __builtin_amdgcn_wmma_f32_16x16x32_bf16(false, ua.v, false, ub.v,
                                                       (short)0, acc[t], false, false);
    }
  }
  const int rowb = rb + wid * 16 + ((lane < 16) ? 0 : 8);
  const int nkt_out = Nc >> 5;
#pragma unroll
  for (int t = 0; t < 4; t++){
    int cc = cb + t * 16 + (lane & 15);
#pragma unroll
    for (int r = 0; r < 8; r++){
      int rr = rowb + r;
      if (rr >= Mr) continue;
      float v = acc[t][r];
      if (bias)  v += bias[cc];
      if (resid) v += resid[(size_t)rr * Nc + cc];
      if (act == 1) v = fmaxf(v, 0.0f);
      else if (act == 2) v = gelu_f(v);
      if (C)  C[(size_t)rr * Nc + cc] = v;
      if (Cf) Cf[((size_t)(rr >> 4) * nkt_out + (cc >> 5)) * 512 + fragA_idx(rr, cc)] = f2bf(v);
    }
  }
}

// ---------------- fused multi-head attention ----------------
// one block per (head, batch); q/k/v in fragment-order LDS (bf16);
// raw scores S (fp32) and probabilities P (bf16, A-fragment order) in global scratch.
// Output written directly in A-fragment bf16 order (Kdim = D_MODEL) for the proj GEMM.
template<int TPAD>
__global__ __launch_bounds__(256) void attn_kernel(
    const float* __restrict__ qkv, unsigned short* __restrict__ of,
    float* __restrict__ Sbase, unsigned short* __restrict__ Pbase, int T)
{
  extern __shared__ unsigned short smem_u16[];
  unsigned short* qs = smem_u16;                 // [NT][2][512] A-order
  unsigned short* ks = qs + TPAD * HDIM;         // [NT][2][512] B-order
  unsigned short* vs = ks + TPAD * HDIM;         // [TPAD/32][4][512] B-order
  const int h = blockIdx.x, b = blockIdx.y;
  const int tid = threadIdx.x, lane = tid & 31, wid = tid >> 5;
  float* S = Sbase + (size_t)(b * HEADS + h) * TPAD * TPAD;
  unsigned short* P = Pbase + (size_t)(b * HEADS + h) * TPAD * TPAD;
  constexpr int NT  = TPAD / 16;
  constexpr int NKC = TPAD / 32;

  // stage q/k/v in fragment order (8 features per thread-group)
  for (int gi = tid; gi < TPAD * (HDIM / 8); gi += 256){
    const int t = gi >> 3, d0 = (gi & 7) << 3;
    float4 q0={0,0,0,0}, q1=q0, k0=q0, k1=q0, v0=q0, v1=q0;
    if (t < T){
      const float* row = qkv + (size_t)(b * T + t) * (3 * D_MODEL) + h * HDIM + d0;
      q0 = *(const float4*)(row);               q1 = *(const float4*)(row + 4);
      k0 = *(const float4*)(row + D_MODEL);     k1 = *(const float4*)(row + D_MODEL + 4);
      v0 = *(const float4*)(row + 2*D_MODEL);   v1 = *(const float4*)(row + 2*D_MODEL + 4);
    }
    {
      uint4 w; unsigned int* wp = (unsigned int*)&w;
      wp[0]=pk2(q0.x,q0.y); wp[1]=pk2(q0.z,q0.w); wp[2]=pk2(q1.x,q1.y); wp[3]=pk2(q1.z,q1.w);
      *(uint4*)&qs[((t >> 4) * 2 + (d0 >> 5)) * 512 + fragA_idx(t, d0)] = w;
    }
    {
      uint4 w; unsigned int* wp = (unsigned int*)&w;
      wp[0]=pk2(k0.x,k0.y); wp[1]=pk2(k0.z,k0.w); wp[2]=pk2(k1.x,k1.y); wp[3]=pk2(k1.z,k1.w);
      *(uint4*)&ks[((t >> 4) * 2 + (d0 >> 5)) * 512 + fragB_idx(t, d0)] = w;
    }
    {
      float vv[8] = {v0.x,v0.y,v0.z,v0.w,v1.x,v1.y,v1.z,v1.w};
      const int cbase = (t >> 5) * 4;
#pragma unroll
      for (int j = 0; j < 8; j++){
        int d = d0 + j;
        vs[(cbase + (d >> 4)) * 512 + fragB_idx(d, t)] = f2bf(vv[j]);
      }
    }
  }
  __syncthreads();

  const float scale = 0.125f;                 // hd^-0.5, hd = 64
  // Phase 1: S = scale * Q @ K^T
  for (int tile = wid; tile < NT * NT; tile += 8){
    const int ti = tile / NT, tj = tile % NT;
    v8f acc = v8f_zero();
#pragma unroll
    for (int kk = 0; kk < 2; kk++){
      BF16x16 ua, ub;
      const uint4* ap = (const uint4*)&qs[(ti * 2 + kk) * 512 + lane * 16];
      ua.q[0] = ap[0]; ua.q[1] = ap[1];
      const uint4* bp = (const uint4*)&ks[(tj * 2 + kk) * 512 + lane * 16];
      ub.q[0] = bp[0]; ub.q[1] = bp[1];
      acc = __builtin_amdgcn_wmma_f32_16x16x32_bf16(false, ua.v, false, ub.v,
                                                    (short)0, acc, false, false);
    }
    const int cc = tj * 16 + (lane & 15);
    const int rowb = ti * 16 + ((lane < 16) ? 0 : 8);
#pragma unroll
    for (int r = 0; r < 8; r++){
      float v = acc[r] * scale;
      if (cc >= T) v = -3.0e38f;              // mask padded keys
      S[(size_t)(rowb + r) * TPAD + cc] = v;
    }
  }
  __threadfence();
  __syncthreads();
  // Phase 2: row softmax in registers -> P (bf16, A-fragment order)
  for (int r = wid; r < TPAD; r += 8){
    const float* srow = S + (size_t)r * TPAD;
    float ev[NKC];
    float mx = -3.0e38f;
#pragma unroll
    for (int j = 0; j < NKC; j++){
      ev[j] = srow[lane + j * 32];
      mx = fmaxf(mx, ev[j]);
    }
    mx = wave_max(mx);
    float sum = 0.f;
#pragma unroll
    for (int j = 0; j < NKC; j++){ ev[j] = __expf(ev[j] - mx); sum += ev[j]; }
    sum = wave_sum(sum);
    const float inv = 1.0f / sum;
#pragma unroll
    for (int j = 0; j < NKC; j++){
      int c = lane + j * 32;
      P[((r >> 4) * NKC + (c >> 5)) * 512 + fragA_idx(r, c)] = f2bf(ev[j] * inv);
    }
  }
  __threadfence();
  __syncthreads();
  // Phase 3: O = P @ V -> A-fragment bf16 (Kdim = D_MODEL) for the proj GEMM
  for (int tile = wid; tile < NT * 4; tile += 8){
    const int ti = tile >> 2, tj = tile & 3;
    v8f acc = v8f_zero();
#pragma unroll
    for (int kk = 0; kk < NKC; kk++){
      BF16x16 ua, ub;
      const uint4* ap = (const uint4*)&P[((size_t)ti * NKC + kk) * 512 + lane * 16];
      ua.q[0] = ap[0]; ua.q[1] = ap[1];
      const uint4* bp = (const uint4*)&vs[(kk * 4 + tj) * 512 + lane * 16];
      ub.q[0] = bp[0]; ub.q[1] = bp[1];
      acc = __builtin_amdgcn_wmma_f32_16x16x32_bf16(false, ua.v, false, ub.v,
                                                    (short)0, acc, false, false);
    }
    const int col = h * HDIM + tj * 16 + (lane & 15);
    const int rowb = ti * 16 + ((lane < 16) ? 0 : 8);
#pragma unroll
    for (int r = 0; r < 8; r++){
      int t = rowb + r;
      if (t < T){
        int row = b * T + t;
        of[((size_t)(row >> 4) * (D_MODEL >> 5) + (col >> 5)) * 512 + fragA_idx(row, col)]
            = f2bf(acc[r]);
      }
    }
  }
}

// ---------------- fused group encoder (mini-PointNet) ----------------
__global__ __launch_bounds__(256) void group_enc_kernel(
    const float* __restrict__ pg, const float* __restrict__ fc1,
    const float* __restrict__ bn1scale, const float* __restrict__ bn1shift,
    const unsigned short* __restrict__ fc2f, const unsigned short* __restrict__ tqf,
    const unsigned short* __restrict__ tkf, unsigned short* __restrict__ featF)
{
  alignas(16) __shared__ unsigned short hF[2 * 4 * 512];   // A-order, 32x128
  alignas(16) __shared__ unsigned short fF[2 * 8 * 512];   // A-order, 32x256
  alignas(16) __shared__ unsigned short qF[2 * 8 * 512];   // A-order, 32x256
  alignas(16) __shared__ unsigned short kF[2 * 8 * 512];   // B-order, 32 cols x 256 k
  __shared__ float ssh[32][33];
  __shared__ float dsh[32];
  const int g = blockIdx.x;
  const int tid = threadIdx.x, lane = tid & 31, wid = tid >> 5;
  const float* P = pg + (size_t)g * K_NN * 3;

  // h = relu(bn(pg @ fc1)), written in A-fragment order (8 channels per group)
  for (int gi = tid; gi < 512; gi += 256){   // 32 rows * 16 col-groups
    const int r = gi >> 4, c0 = (gi & 15) << 3;
    const float x0 = P[r*3], x1 = P[r*3+1], x2 = P[r*3+2];
    float hv[8];
#pragma unroll
    for (int j = 0; j < 8; j++){
      const int c = c0 + j;
      float v = x0*fc1[c] + x1*fc1[128+c] + x2*fc1[256+c];
      v = v * bn1scale[c] + bn1shift[c];
      hv[j] = fmaxf(v, 0.0f);
    }
    uint4 w; unsigned int* wp = (unsigned int*)&w;
    wp[0]=pk2(hv[0],hv[1]); wp[1]=pk2(hv[2],hv[3]); wp[2]=pk2(hv[4],hv[5]); wp[3]=pk2(hv[6],hv[7]);
    *(uint4*)&hF[((r >> 4) * 4 + (c0 >> 5)) * 512 + fragA_idx(r, c0)] = w;
  }
  __syncthreads();

  // f = h @ fc2  (32x128 @ 128x256)
  for (int tile = wid; tile < 32; tile += 8){
    const int ti = tile >> 4, tj = tile & 15;
    v8f acc = v8f_zero();
#pragma unroll
    for (int kk = 0; kk < 4; kk++){
      BF16x16 ua, ub;
      const uint4* ap = (const uint4*)&hF[(ti * 4 + kk) * 512 + lane * 16];
      ua.q[0] = ap[0]; ua.q[1] = ap[1];
      const uint4* bp = (const uint4*)(fc2f + (size_t)(kk * 16 + tj) * 512 + lane * 16);
      ub.q[0] = bp[0]; ub.q[1] = bp[1];
      acc = __builtin_amdgcn_wmma_f32_16x16x32_bf16(false, ua.v, false, ub.v,
                                                    (short)0, acc, false, false);
    }
    const int cc = tj * 16 + (lane & 15);
    const int rowb = ti * 16 + ((lane < 16) ? 0 : 8);
#pragma unroll
    for (int r = 0; r < 8; r++){
      const int row = rowb + r;
      fF[((row >> 4) * 8 + (cc >> 5)) * 512 + fragA_idx(row, cc)] = f2bf(acc[r]);
    }
  }
  __syncthreads();

  // q = f @ tq ; k = f @ tk  (32x256 @ 256x256 each)
  for (int tile = wid; tile < 64; tile += 8){
    const int sel = tile >> 5, t2 = tile & 31;
    const int ti = t2 >> 4, tj = t2 & 15;
    const unsigned short* W = sel ? tkf : tqf;
    v8f acc = v8f_zero();
#pragma unroll
    for (int kk = 0; kk < 8; kk++){
      BF16x16 ua, ub;
      const uint4* ap = (const uint4*)&fF[(ti * 8 + kk) * 512 + lane * 16];
      ua.q[0] = ap[0]; ua.q[1] = ap[1];
      const uint4* bp = (const uint4*)(W + (size_t)(kk * 16 + tj) * 512 + lane * 16);
      ub.q[0] = bp[0]; ub.q[1] = bp[1];
      acc = __builtin_amdgcn_wmma_f32_16x16x32_bf16(false, ua.v, false, ub.v,
                                                    (short)0, acc, false, false);
    }
    const int cc = tj * 16 + (lane & 15);
    const int rowb = ti * 16 + ((lane < 16) ? 0 : 8);
    if (sel == 0){
#pragma unroll
      for (int r = 0; r < 8; r++){
        const int row = rowb + r;
        qF[((row >> 4) * 8 + (cc >> 5)) * 512 + fragA_idx(row, cc)] = f2bf(acc[r]);
      }
    } else {
#pragma unroll
      for (int r = 0; r < 8; r++){
        const int row = rowb + r;
        kF[((row >> 4) * 8 + (cc >> 5)) * 512 + fragB_idx(row, cc)] = f2bf(acc[r]);
      }
    }
  }
  __syncthreads();

  // s = q @ k^T (32x32, K=256)
  for (int tile = wid; tile < 4; tile += 8){
    const int ti = tile >> 1, tj = tile & 1;
    v8f acc = v8f_zero();
#pragma unroll
    for (int kk = 0; kk < 8; kk++){
      BF16x16 ua, ub;
      const uint4* ap = (const uint4*)&qF[(ti * 8 + kk) * 512 + lane * 16];
      ua.q[0] = ap[0]; ua.q[1] = ap[1];
      const uint4* bp = (const uint4*)&kF[(tj * 8 + kk) * 512 + lane * 16];
      ub.q[0] = bp[0]; ub.q[1] = bp[1];
      acc = __builtin_amdgcn_wmma_f32_16x16x32_bf16(false, ua.v, false, ub.v,
                                                    (short)0, acc, false, false);
    }
    const int cc = tj * 16 + (lane & 15);
    const int rowb = ti * 16 + ((lane < 16) ? 0 : 8);
#pragma unroll
    for (int r = 0; r < 8; r++) ssh[rowb + r][cc] = acc[r];
  }
  __syncthreads();

  // softmax diagonal (per row)
  if (tid < 32){
    const int r = tid;
    float mx = -3.0e38f;
    for (int c = 0; c < 32; c++) mx = fmaxf(mx, ssh[r][c]);
    float sum = 0.f;
    for (int c = 0; c < 32; c++) sum += __expf(ssh[r][c] - mx);
    float d = __expf(ssh[r][r] - mx) / sum;
    dsh[r] = d * (1.0f / (1.0f + 1e-8f));
  }
  __syncthreads();

  // fa = diag * f ; feat = [mean_k, max_k] -> A-fragment bf16 (rows = group id, K = 512)
  {
    const int c = tid;  // 0..255
    float s = 0.f, m = -3.0e38f;
    for (int r = 0; r < 32; r++){
      float v = dsh[r] * bf2f(fF[((r >> 4) * 8 + (c >> 5)) * 512 + fragA_idx(r, c)]);
      s += v; m = fmaxf(m, v);
    }
    const int c2 = c + 256;
    featF[((size_t)(g >> 4) * 16 + (c >> 5)) * 512 + fragA_idx(g, c)]   = f2bf(s * (1.0f/32.0f));
    featF[((size_t)(g >> 4) * 16 + (c2 >> 5)) * 512 + fragA_idx(g, c2)] = f2bf(m);
  }
}

// ---------------- FPS ----------------
__global__ __launch_bounds__(256) void fps_kernel(const float* __restrict__ xyz, int* __restrict__ cidx)
{
  __shared__ float dist[N_PTS];
  __shared__ float rv[256];
  __shared__ int   ri[256];
  __shared__ int   s_last;
  const int b = blockIdx.x, tid = threadIdx.x;
  const float* P = xyz + (size_t)b * N_PTS * 3;
  for (int n = tid; n < N_PTS; n += 256) dist[n] = 1.0e10f;
  if (tid == 0){ cidx[b * G_GRP] = 0; s_last = 0; }
  __syncthreads();
  for (int s = 1; s < G_GRP; s++){
    const int last = s_last;
    const float px = P[last*3], py = P[last*3+1], pz = P[last*3+2];
    float bv = -1.0f; int bidx = 0;
    for (int n = tid; n < N_PTS; n += 256){
      float dx = P[n*3]-px, dy = P[n*3+1]-py, dz = P[n*3+2]-pz;
      float d = dx*dx + dy*dy + dz*dz;
      float dn = fminf(dist[n], d);
      dist[n] = dn;
      if (dn > bv){ bv = dn; bidx = n; }
    }
    rv[tid] = bv; ri[tid] = bidx;
    __syncthreads();
    for (int st = 128; st > 0; st >>= 1){
      if (tid < st){
        float ov = rv[tid + st]; int oi = ri[tid + st];
        if (ov > rv[tid] || (ov == rv[tid] && oi < ri[tid])){ rv[tid] = ov; ri[tid] = oi; }
      }
      __syncthreads();
    }
    if (tid == 0){ cidx[b * G_GRP + s] = ri[0]; s_last = ri[0]; }
    __syncthreads();
  }
}

// ---------------- KNN + grouping ----------------
__global__ __launch_bounds__(256) void knn_group_kernel(
    const float* __restrict__ xyz, const float* __restrict__ center, float* __restrict__ pg)
{
  const int idx = blockIdx.x * 256 + threadIdx.x;
  if (idx >= B_SZ * G_GRP) return;
  const int b = idx / G_GRP;
  const float cx = center[(size_t)idx*3], cy = center[(size_t)idx*3+1], cz = center[(size_t)idx*3+2];
  float bd[K_NN]; int bi[K_NN];
#pragma unroll
  for (int k = 0; k < K_NN; k++){ bd[k] = 3.0e38f; bi[k] = 0; }
  const float* P = xyz + (size_t)b * N_PTS * 3;
  for (int n = 0; n < N_PTS; n++){
    float dx = P[n*3]-cx, dy = P[n*3+1]-cy, dz = P[n*3+2]-cz;
    float d = dx*dx + dy*dy + dz*dz;
    if (d < bd[K_NN-1]){
      int j = K_NN - 1;
      while (j > 0 && bd[j-1] > d){ bd[j] = bd[j-1]; bi[j] = bi[j-1]; j--; }
      bd[j] = d; bi[j] = n;
    }
  }
  float* out = pg + (size_t)idx * K_NN * 3;
  for (int k = 0; k < K_NN; k++){
    int n = bi[k];
    out[k*3]   = P[n*3]   - cx;
    out[k*3+1] = P[n*3+1] - cy;
    out[k*3+2] = P[n*3+2] - cz;
  }
}

// ---------------- BN helpers ----------------
__global__ void bn1_stats_kernel(const float* __restrict__ pg, const float* __restrict__ fc1,
                                 float* __restrict__ stats)
{
  const int c = threadIdx.x & 127, rl = threadIdx.x >> 7;
  const int r0 = blockIdx.x * 256;
  float s = 0.f, s2 = 0.f;
  for (int r = r0 + rl; r < r0 + 256; r += 2){
    const float* p = pg + (size_t)r * 3;
    float hv = p[0]*fc1[c] + p[1]*fc1[128+c] + p[2]*fc1[256+c];
    s += hv; s2 += hv*hv;
  }
  atomicAdd(&stats[c], s);
  atomicAdd(&stats[128 + c], s2);
}

__global__ void bn_stats_kernel(const float* __restrict__ X, int Mr, int C, float* __restrict__ stats)
{
  const int c = blockIdx.x * blockDim.x + threadIdx.x;
  if (c >= C) return;
  const int rows_per = (Mr + gridDim.y - 1) / gridDim.y;
  const int r0 = blockIdx.y * rows_per;
  int r1 = r0 + rows_per; if (r1 > Mr) r1 = Mr;
  float s = 0.f, s2 = 0.f;
  for (int r = r0; r < r1; r++){ float v = X[(size_t)r * C + c]; s += v; s2 += v*v; }
  atomicAdd(&stats[c], s);
  atomicAdd(&stats[C + c], s2);
}

__global__ void bn_finalize_kernel(const float* __restrict__ stats,
                                   const float* __restrict__ g, const float* __restrict__ b,
                                   float* __restrict__ scale, float* __restrict__ shift,
                                   int C, float invcnt)
{
  const int c = blockIdx.x * blockDim.x + threadIdx.x;
  if (c >= C) return;
  float mean = stats[c] * invcnt;
  float var  = stats[C + c] * invcnt - mean * mean;
  float rs = rsqrtf(var + 1e-5f);
  scale[c] = g[c] * rs;
  shift[c] = b[c] - mean * g[c] * rs;
}

// relu(bn(x)) -> A-fragment bf16
__global__ void bn_apply_relu_kernel(const float* __restrict__ X, const float* __restrict__ scale,
                                     const float* __restrict__ shift,
                                     unsigned short* __restrict__ outF, int n, int C)
{
  const int i = blockIdx.x * blockDim.x + threadIdx.x;
  if (i >= n) return;
  const int r = i / C, c = i % C;
  float v = fmaxf(X[i] * scale[c] + shift[c], 0.0f);
  outF[((size_t)(r >> 4) * (C >> 5) + (c >> 5)) * 512 + fragA_idx(r, c)] = f2bf(v);
}

// ---------------- LayerNorm: fp32 in -> A-fragment bf16 out ----------------
__global__ __launch_bounds__(128) void ln_kernel(const float* __restrict__ X,
    const float* __restrict__ gg, const float* __restrict__ bb,
    unsigned short* __restrict__ Yf, int C)
{
  const int r = blockIdx.x, tid = threadIdx.x, lane = tid & 31, wid = tid >> 5;
  const float* x = X + (size_t)r * C;
  __shared__ float red1[4], red2[4];
  float s = 0.f;
  for (int c = tid; c < C; c += 128) s += x[c];
  s = wave_sum(s);
  if (lane == 0) red1[wid] = s;
  __syncthreads();
  const float mean = (red1[0]+red1[1]+red1[2]+red1[3]) / (float)C;
  float v = 0.f;
  for (int c = tid; c < C; c += 128){ float d = x[c] - mean; v += d*d; }
  v = wave_sum(v);
  if (lane == 0) red2[wid] = v;
  __syncthreads();
  const float rs = rsqrtf((red2[0]+red2[1]+red2[2]+red2[3]) / (float)C + 1e-5f);
  const int nkt = C >> 5;
  for (int c = tid; c < C; c += 128){
    float y = (x[c]-mean)*rs*gg[c] + bb[c];
    Yf[((size_t)(r >> 4) * nkt + (c >> 5)) * 512 + fragA_idx(r, c)] = f2bf(y);
  }
}

__global__ __launch_bounds__(128) void final_ln_kernel(const float* __restrict__ X,
    const float* __restrict__ gg, const float* __restrict__ bb, float* __restrict__ out)
{
  const int rr = blockIdx.x;                  // 0..B*M_MASK-1
  const int b = rr / M_MASK, j = rr % M_MASK;
  const int tid = threadIdx.x, lane = tid & 31, wid = tid >> 5;
  const float* x = X + (size_t)(b * G_GRP + N_VIS + j) * D_MODEL;
  float* y = out + (size_t)rr * D_MODEL;
  __shared__ float red1[4], red2[4];
  float s = 0.f;
  for (int c = tid; c < D_MODEL; c += 128) s += x[c];
  s = wave_sum(s);
  if (lane == 0) red1[wid] = s;
  __syncthreads();
  const float mean = (red1[0]+red1[1]+red1[2]+red1[3]) / (float)D_MODEL;
  float v = 0.f;
  for (int c = tid; c < D_MODEL; c += 128){ float d = x[c] - mean; v += d*d; }
  v = wave_sum(v);
  if (lane == 0) red2[wid] = v;
  __syncthreads();
  const float rs = rsqrtf((red2[0]+red2[1]+red2[2]+red2[3]) / (float)D_MODEL + 1e-5f);
  for (int c = tid; c < D_MODEL; c += 128) y[c] = (x[c]-mean)*rs*gg[c] + bb[c];
}

// ---------------- small elementwise / gather kernels ----------------
__global__ void zero_f_kernel(float* p, int n){
  int i = blockIdx.x * blockDim.x + threadIdx.x; if (i < n) p[i] = 0.f;
}
__global__ void add_kernel(float* __restrict__ out, const float* __restrict__ a,
                           const float* __restrict__ b, int n){
  int i = blockIdx.x * blockDim.x + threadIdx.x; if (i < n) out[i] = a[i] + b[i];
}
// K=3 GEMM + bias + gelu -> A-fragment bf16
__global__ void k3_gemm_kernel(const float* __restrict__ X, const float* __restrict__ W,
                               const float* __restrict__ bias, unsigned short* __restrict__ Yf,
                               int Mr, int C, int act){
  int i = blockIdx.x * blockDim.x + threadIdx.x;
  if (i >= Mr * C) return;
  int r = i / C, c = i % C;
  const float* x = X + (size_t)r * 3;
  float v = x[0]*W[c] + x[1]*W[C+c] + x[2]*W[2*C+c];
  if (bias) v += bias[c];
  if (act == 2) v = gelu_f(v);
  Yf[((size_t)(r >> 4) * (C >> 5) + (c >> 5)) * 512 + fragA_idx(r, c)] = f2bf(v);
}
__global__ void gather_center_kernel(const float* __restrict__ xyz, const int* __restrict__ cidx,
                                     float* __restrict__ center){
  int i = blockIdx.x * blockDim.x + threadIdx.x;
  if (i >= B_SZ * G_GRP * 3) return;
  int row = i / 3, c = i % 3;
  int b = row / G_GRP;
  center[i] = xyz[((size_t)b * N_PTS + cidx[row]) * 3 + c];
}
__global__ void partition_kernel(const int* __restrict__ mask, int* __restrict__ vis, int* __restrict__ msk){
  const int b = blockIdx.x;
  if (threadIdx.x != 0) return;
  int nv = 0, nm = 0;
  for (int g = 0; g < G_GRP; g++){
    if (mask[b * G_GRP + g] == 0) vis[b * N_VIS + (nv++)] = g;
    else                          msk[b * M_MASK + (nm++)] = g;
  }
}
__global__ void gather_tokens_kernel(const float* __restrict__ tokens, const int* __restrict__ vis,
                                     float* __restrict__ xv){
  int i = blockIdx.x * blockDim.x + threadIdx.x;
  if (i >= M_ENC * D_MODEL) return;
  int row = i / D_MODEL, c = i % D_MODEL;
  int b = row / N_VIS, t = row % N_VIS;
  int g = vis[b * N_VIS + t];
  xv[i] = tokens[((size_t)(b * G_GRP + g)) * D_MODEL + c];
}
__global__ void build_cf_kernel(const float* __restrict__ center, const int* __restrict__ vis,
                                const int* __restrict__ msk, float* __restrict__ cf){
  int i = blockIdx.x * blockDim.x + threadIdx.x;
  if (i >= B_SZ * G_GRP * 3) return;
  int row = i / 3, c = i % 3;
  int b = row / G_GRP, t = row % G_GRP;
  int g = (t < N_VIS) ? vis[b * N_VIS + t] : msk[b * M_MASK + (t - N_VIS)];
  cf[i] = center[((size_t)(b * G_GRP + g)) * 3 + c];
}
__global__ void copy_visc_kernel(const float* __restrict__ cf, float* __restrict__ visc){
  int i = blockIdx.x * blockDim.x + threadIdx.x;
  if (i >= M_ENC * 3) return;
  int row = i / 3, c = i % 3;
  int b = row / N_VIS, t = row % N_VIS;
  visc[i] = cf[((size_t)(b * G_GRP + t)) * 3 + c];
}
__global__ void dec_build_kernel(const float* __restrict__ xd, const float* __restrict__ mask_token,
                                 float* __restrict__ xf){
  int i = blockIdx.x * blockDim.x + threadIdx.x;
  if (i >= M_DEC * D_MODEL) return;
  int row = i / D_MODEL, c = i % D_MODEL;
  int b = row / G_GRP, t = row % G_GRP;
  xf[i] = (t < N_VIS) ? xd[((size_t)(b * N_VIS + t)) * D_MODEL + c] : mask_token[c];
}

// ---------------- orchestration ----------------
extern "C" void kernel_launch(void* const* d_in, const int* in_sizes, int n_in,
                              void* d_out, int out_size, void* d_ws, size_t ws_size,
                              hipStream_t stream)
{
  (void)in_sizes; (void)n_in; (void)out_size; (void)ws_size;
  const float* xyz        = (const float*)d_in[0];
  const int*   maskp      = (const int*)  d_in[1];
  const float* enc_fc1_w  = (const float*)d_in[2];
  const float* enc_bn1_g  = (const float*)d_in[3];
  const float* enc_bn1_b  = (const float*)d_in[4];
  const float* enc_fc2_w  = (const float*)d_in[5];
  const float* enc_tq_w   = (const float*)d_in[6];
  const float* enc_tk_w   = (const float*)d_in[7];
  const float* enc_sc1_w  = (const float*)d_in[8];
  const float* enc_bn2_g  = (const float*)d_in[9];
  const float* enc_bn2_b  = (const float*)d_in[10];
  const float* enc_sc2_w  = (const float*)d_in[11];
  const float* pe1_w      = (const float*)d_in[12];
  const float* pe1_b      = (const float*)d_in[13];
  const float* pe2_w      = (const float*)d_in[14];
  const float* pe2_b      = (const float*)d_in[15];
  const float* blk_ln1_g  = (const float*)d_in[16];
  const float* blk_ln1_b  = (const float*)d_in[17];
  const float* blk_qkv_w  = (const float*)d_in[18];
  const float* blk_proj_w = (const float*)d_in[19];
  const float* blk_proj_b = (const float*)d_in[20];
  const float* blk_ln2_g  = (const float*)d_in[21];
  const float* blk_ln2_b  = (const float*)d_in[22];
  const float* blk_fc1_w  = (const float*)d_in[23];
  const float* blk_fc1_b  = (const float*)d_in[24];
  const float* blk_fc2_w  = (const float*)d_in[25];
  const float* blk_fc2_b  = (const float*)d_in[26];
  const float* norm_g     = (const float*)d_in[27];
  const float* norm_b     = (const float*)d_in[28];
  const float* de_w       = (const float*)d_in[29];
  const float* de_b       = (const float*)d_in[30];
  const float* mask_token = (const float*)d_in[31];
  const float* dpe1_w     = (const float*)d_in[32];
  const float* dpe1_b     = (const float*)d_in[33];
  const float* dpe2_w     = (const float*)d_in[34];
  const float* dpe2_b     = (const float*)d_in[35];
  const float* dec_ln1_g  = (const float*)d_in[36];
  const float* dec_ln1_b  = (const float*)d_in[37];
  const float* dec_qkv_w  = (const float*)d_in[38];
  const float* dec_proj_w = (const float*)d_in[39];
  const float* dec_proj_b = (const float*)d_in[40];
  const float* dec_ln2_g  = (const float*)d_in[41];
  const float* dec_ln2_b  = (const float*)d_in[42];
  const float* dec_fc1_w  = (const float*)d_in[43];
  const float* dec_fc1_b  = (const float*)d_in[44];
  const float* dec_fc2_w  = (const float*)d_in[45];
  const float* dec_fc2_b  = (const float*)d_in[46];
  const float* dnorm_g    = (const float*)d_in[47];
  const float* dnorm_b    = (const float*)d_in[48];
  float* outp = (float*)d_out;

  // workspace allocator
  char* wsB = (char*)d_ws;
  size_t off = 0;
  auto alloc = [&](size_t bytes) -> void* {
    void* p = wsB + off;
    off += (bytes + 255) & ~(size_t)255;
    return p;
  };
  const int M_ENC_PAD = 3328;   // M_ENC padded to 128 rows (gemm over-read safety)
  int*   cidx    = (int*)  alloc((size_t)B_SZ*G_GRP*4);
  float* center  = (float*)alloc((size_t)B_SZ*G_GRP*3*4);
  float* pg      = (float*)alloc((size_t)B_SZ*G_GRP*K_NN*3*4);
  float* bn1st   = (float*)alloc(256*4);
  float* bn1sc   = (float*)alloc(128*4);
  float* bn1sh   = (float*)alloc(128*4);
  float* bn2st   = (float*)alloc(1024*4);
  float* bn2sc   = (float*)alloc(512*4);
  float* bn2sh   = (float*)alloc(512*4);
  float* tokens  = (float*)alloc((size_t)M_DEC*D_MODEL*4);
  int*   visidx  = (int*)  alloc((size_t)B_SZ*N_VIS*4);
  int*   mskidx  = (int*)  alloc((size_t)B_SZ*M_MASK*4);
  float* cf      = (float*)alloc((size_t)B_SZ*G_GRP*3*4);
  float* visc    = (float*)alloc((size_t)M_ENC*3*4);
  float* pos     = (float*)alloc((size_t)M_ENC*D_MODEL*4);
  float* dpos    = (float*)alloc((size_t)M_DEC*D_MODEL*4);
  float* xv      = (float*)alloc((size_t)M_ENC*D_MODEL*4);
  float* xd      = (float*)alloc((size_t)M_ENC*D_MODEL*4);
  float* xf      = (float*)alloc((size_t)M_DEC*D_MODEL*4);
  float* xin     = (float*)alloc((size_t)M_DEC*D_MODEL*4);
  float* t1      = (float*)alloc((size_t)M_DEC*512*4);
  float* qkvb    = (float*)alloc((size_t)M_DEC*3*D_MODEL*4);
  float* Sws     = (float*)alloc((size_t)B_SZ*HEADS*256*256*4);
  unsigned short* Pws = (unsigned short*)alloc((size_t)B_SZ*HEADS*256*256*2);
  // bf16 A-fragment activation buffers (rows padded to multiple of 128)
  unsigned short* hbufF  = (unsigned short*)alloc((size_t)M_DEC*D_MODEL*2);
  unsigned short* obufF  = (unsigned short*)alloc((size_t)M_DEC*D_MODEL*2);
  unsigned short* f1F    = (unsigned short*)alloc((size_t)M_DEC*4*D_MODEL*2);
  unsigned short* featF  = (unsigned short*)alloc((size_t)M_DEC*512*2);
  unsigned short* t1F    = (unsigned short*)alloc((size_t)M_DEC*512*2);
  unsigned short* poshF  = (unsigned short*)alloc((size_t)M_ENC_PAD*128*2);
  unsigned short* dposhF = (unsigned short*)alloc((size_t)M_DEC*128*2);
  // packed bf16 weights (fragment order)
  unsigned short* fc2f  = (unsigned short*)alloc((size_t)128*256*2);
  unsigned short* tqf   = (unsigned short*)alloc((size_t)256*256*2);
  unsigned short* tkf   = (unsigned short*)alloc((size_t)256*256*2);
  unsigned short* sc1f  = (unsigned short*)alloc((size_t)512*512*2);
  unsigned short* sc2f  = (unsigned short*)alloc((size_t)512*384*2);
  unsigned short* pe2f  = (unsigned short*)alloc((size_t)128*384*2);
  unsigned short* dpe2f = (unsigned short*)alloc((size_t)128*384*2);
  unsigned short* dewf  = (unsigned short*)alloc((size_t)384*384*2);
  unsigned short* qkvwf = (unsigned short*)alloc((size_t)12*384*1152*2);
  unsigned short* prjwf = (unsigned short*)alloc((size_t)12*384*384*2);
  unsigned short* f1wf  = (unsigned short*)alloc((size_t)12*384*1536*2);
  unsigned short* f2wf  = (unsigned short*)alloc((size_t)12*1536*384*2);
  unsigned short* dqkvf = (unsigned short*)alloc((size_t)4*384*1152*2);
  unsigned short* dprjf = (unsigned short*)alloc((size_t)4*384*384*2);
  unsigned short* df1f  = (unsigned short*)alloc((size_t)4*384*1536*2);
  unsigned short* df2f  = (unsigned short*)alloc((size_t)4*1536*384*2);

  auto cdiv = [](int a, int b){ return (a + b - 1) / b; };
  auto pack = [&](const float* W, int K, int N, unsigned short* out){
    pack_wB_kernel<<<cdiv(K*N, 256), 256, 0, stream>>>(W, K, N, out);
  };
  auto gemm = [&](const unsigned short* Af, const unsigned short* Bf, float* C,
                  unsigned short* Cf, const float* bias, const float* resid,
                  int Mr, int Nc, int Kd, int act){
    dim3 g(cdiv(Nc, 64), cdiv(Mr, 128));
    gemm_bf16_kernel<<<g, 256, 0, stream>>>(Af, Bf, C, Cf, bias, resid, Mr, Nc, Kd, act);
  };

  // ---- stage 0: pre-swizzle all weights to bf16 fragment order ----
  pack(enc_fc2_w, 128, 256, fc2f);
  pack(enc_tq_w, 256, 256, tqf);
  pack(enc_tk_w, 256, 256, tkf);
  pack(enc_sc1_w, 512, 512, sc1f);
  pack(enc_sc2_w, 512, 384, sc2f);
  pack(pe2_w, 128, 384, pe2f);
  pack(dpe2_w, 128, 384, dpe2f);
  pack(de_w, 384, 384, dewf);
  for (int i = 0; i < 12; i++){
    pack(blk_qkv_w  + (size_t)i*384*1152, 384, 1152, qkvwf + (size_t)i*384*1152);
    pack(blk_proj_w + (size_t)i*384*384,  384, 384,  prjwf + (size_t)i*384*384);
    pack(blk_fc1_w  + (size_t)i*384*1536, 384, 1536, f1wf  + (size_t)i*384*1536);
    pack(blk_fc2_w  + (size_t)i*1536*384, 1536, 384, f2wf  + (size_t)i*1536*384);
  }
  for (int i = 0; i < 4; i++){
    pack(dec_qkv_w  + (size_t)i*384*1152, 384, 1152, dqkvf + (size_t)i*384*1152);
    pack(dec_proj_w + (size_t)i*384*384,  384, 384,  dprjf + (size_t)i*384*384);
    pack(dec_fc1_w  + (size_t)i*384*1536, 384, 1536, df1f  + (size_t)i*384*1536);
    pack(dec_fc2_w  + (size_t)i*1536*384, 1536, 384, df2f  + (size_t)i*1536*384);
  }

  // ---- stage 1: FPS -> centers -> KNN grouping ----
  fps_kernel<<<B_SZ, 256, 0, stream>>>(xyz, cidx);
  gather_center_kernel<<<cdiv(B_SZ*G_GRP*3, 256), 256, 0, stream>>>(xyz, cidx, center);
  knn_group_kernel<<<cdiv(B_SZ*G_GRP, 256), 256, 0, stream>>>(xyz, center, pg);

  // ---- stage 2: BN1 stats + fused group encoder ----
  zero_f_kernel<<<1, 256, 0, stream>>>(bn1st, 256);
  bn1_stats_kernel<<<1024, 256, 0, stream>>>(pg, enc_fc1_w, bn1st);
  bn_finalize_kernel<<<1, 128, 0, stream>>>(bn1st, enc_bn1_g, enc_bn1_b, bn1sc, bn1sh,
                                            128, 1.0f / (float)(B_SZ*G_GRP*K_NN));
  group_enc_kernel<<<B_SZ*G_GRP, 256, 0, stream>>>(pg, enc_fc1_w, bn1sc, bn1sh,
                                                   fc2f, tqf, tkf, featF);

  // ---- stage 3: token head (feat -> tokens) ----
  gemm(featF, sc1f, t1, nullptr, nullptr, nullptr, M_DEC, 512, 512, 0);
  zero_f_kernel<<<4, 256, 0, stream>>>(bn2st, 1024);
  bn_stats_kernel<<<dim3(2, 32), 256, 0, stream>>>(t1, M_DEC, 512, bn2st);
  bn_finalize_kernel<<<2, 256, 0, stream>>>(bn2st, enc_bn2_g, enc_bn2_b, bn2sc, bn2sh,
                                            512, 1.0f / (float)M_DEC);
  bn_apply_relu_kernel<<<cdiv(M_DEC*512, 256), 256, 0, stream>>>(t1, bn2sc, bn2sh, t1F,
                                                                 M_DEC*512, 512);
  gemm(t1F, sc2f, tokens, nullptr, nullptr, nullptr, M_DEC, D_MODEL, 512, 0);

  // ---- stage 4: mask partition, gathers, positional embeddings ----
  partition_kernel<<<B_SZ, 32, 0, stream>>>(maskp, visidx, mskidx);
  gather_tokens_kernel<<<cdiv(M_ENC*D_MODEL, 256), 256, 0, stream>>>(tokens, visidx, xv);
  build_cf_kernel<<<cdiv(B_SZ*G_GRP*3, 256), 256, 0, stream>>>(center, visidx, mskidx, cf);
  copy_visc_kernel<<<cdiv(M_ENC*3, 256), 256, 0, stream>>>(cf, visc);
  k3_gemm_kernel<<<cdiv(M_ENC*128, 256), 256, 0, stream>>>(visc, pe1_w, pe1_b, poshF, M_ENC, 128, 2);
  gemm(poshF, pe2f, pos, nullptr, pe2_b, nullptr, M_ENC, D_MODEL, 128, 0);
  k3_gemm_kernel<<<cdiv(M_DEC*128, 256), 256, 0, stream>>>(cf, dpe1_w, dpe1_b, dposhF, M_DEC, 128, 2);
  gemm(dposhF, dpe2f, dpos, nullptr, dpe2_b, nullptr, M_DEC, D_MODEL, 128, 0);

  // ---- stage 5: 12-layer encoder (T=103, pad 128) ----
  for (int i = 0; i < 12; i++){
    const unsigned short* Wqkv = qkvwf + (size_t)i*384*1152;
    const unsigned short* Wprj = prjwf + (size_t)i*384*384;
    const unsigned short* W1   = f1wf  + (size_t)i*384*1536;
    const unsigned short* W2   = f2wf  + (size_t)i*1536*384;
    const float* bprj = blk_proj_b + (size_t)i*D_MODEL;
    const float* b1   = blk_fc1_b + (size_t)i*4*D_MODEL;
    const float* b2   = blk_fc2_b + (size_t)i*D_MODEL;
    add_kernel<<<cdiv(M_ENC*D_MODEL, 256), 256, 0, stream>>>(xin, xv, pos, M_ENC*D_MODEL);
    ln_kernel<<<M_ENC, 128, 0, stream>>>(xin, blk_ln1_g + i*D_MODEL, blk_ln1_b + i*D_MODEL,
                                         hbufF, D_MODEL);
    gemm(hbufF, Wqkv, qkvb, nullptr, nullptr, nullptr, M_ENC, 3*D_MODEL, D_MODEL, 0);
    attn_kernel<128><<<dim3(HEADS, B_SZ), 256, 3*128*HDIM*sizeof(unsigned short), stream>>>(
        qkvb, obufF, Sws, Pws, N_VIS);
    gemm(obufF, Wprj, xv, nullptr, bprj, xin, M_ENC, D_MODEL, D_MODEL, 0);
    ln_kernel<<<M_ENC, 128, 0, stream>>>(xv, blk_ln2_g + i*D_MODEL, blk_ln2_b + i*D_MODEL,
                                         hbufF, D_MODEL);
    gemm(hbufF, W1, nullptr, f1F, b1, nullptr, M_ENC, 4*D_MODEL, D_MODEL, 2);
    gemm(f1F, W2, xv, nullptr, b2, xv, M_ENC, D_MODEL, 4*D_MODEL, 0);
  }
  ln_kernel<<<M_ENC, 128, 0, stream>>>(xv, norm_g, norm_b, hbufF, D_MODEL);
  gemm(hbufF, dewf, xd, nullptr, de_b, nullptr, M_ENC, D_MODEL, D_MODEL, 0);

  // ---- stage 6: decoder input + 4-layer decoder (T=256) ----
  dec_build_kernel<<<cdiv(M_DEC*D_MODEL, 256), 256, 0, stream>>>(xd, mask_token, xf);
  for (int i = 0; i < 4; i++){
    const unsigned short* Wqkv = dqkvf + (size_t)i*384*1152;
    const unsigned short* Wprj = dprjf + (size_t)i*384*384;
    const unsigned short* W1   = df1f  + (size_t)i*384*1536;
    const unsigned short* W2   = df2f  + (size_t)i*1536*384;
    const float* bprj = dec_proj_b + (size_t)i*D_MODEL;
    const float* b1   = dec_fc1_b + (size_t)i*4*D_MODEL;
    const float* b2   = dec_fc2_b + (size_t)i*D_MODEL;
    add_kernel<<<cdiv(M_DEC*D_MODEL, 256), 256, 0, stream>>>(xin, xf, dpos, M_DEC*D_MODEL);
    ln_kernel<<<M_DEC, 128, 0, stream>>>(xin, dec_ln1_g + i*D_MODEL, dec_ln1_b + i*D_MODEL,
                                         hbufF, D_MODEL);
    gemm(hbufF, Wqkv, qkvb, nullptr, nullptr, nullptr, M_DEC, 3*D_MODEL, D_MODEL, 0);
    attn_kernel<256><<<dim3(HEADS, B_SZ), 256, 3*256*HDIM*sizeof(unsigned short), stream>>>(
        qkvb, obufF, Sws, Pws, G_GRP);
    gemm(obufF, Wprj, xf, nullptr, bprj, xin, M_DEC, D_MODEL, D_MODEL, 0);
    ln_kernel<<<M_DEC, 128, 0, stream>>>(xf, dec_ln2_g + i*D_MODEL, dec_ln2_b + i*D_MODEL,
                                         hbufF, D_MODEL);
    gemm(hbufF, W1, nullptr, f1F, b1, nullptr, M_DEC, 4*D_MODEL, D_MODEL, 2);
    gemm(f1F, W2, xf, nullptr, b2, xf, M_DEC, D_MODEL, 4*D_MODEL, 0);
  }

  // ---- stage 7: final LN + slice masked tokens ----
  final_ln_kernel<<<B_SZ*M_MASK, 128, 0, stream>>>(xf, dnorm_g, dnorm_b, outp);
}
